// GCN_73237782331661
// MI455X (gfx1250) — matrix operand order repaired and checked by reading, not compiled, not run on hardware
//
#include <hip/hip_runtime.h>
#include <hip/hip_bf16.h>

// ---------------------------------------------------------------------------
// Fused 7-layer GCN + classifier for MI455X (gfx1250), wave32 + WMMA f16.
//
// All contractions are skinny GEMMs with batch-shared B operands:
//   aggregation : rows=(sample,feat),  K=nodes(18->32),  B = A^T (LDS image)
//   per-node fc : rows=(sample,node),  K=feats(->32),    B = W^T (LDS image)
//   classifier  : rows=samples,        K=576 (18x32),    B = Wout (global/L2)
// X tiles are DMA'd into LDS by the Tensor Data Mover when available.
// C-matrix writebacks exploit layout linearity: stage A uses ds_store_b128,
// stage B uses stride-increment addressing (no per-element div/mul).
// ---------------------------------------------------------------------------

typedef _Float16 half_t;
typedef __attribute__((ext_vector_type(16))) _Float16 v16h;
typedef __attribute__((ext_vector_type(8)))  _Float16 v8h;
typedef __attribute__((ext_vector_type(8)))  float    v8f;
typedef __attribute__((ext_vector_type(4)))  unsigned int uint4v;
typedef __attribute__((ext_vector_type(8)))  int int8v;
typedef __attribute__((ext_vector_type(4)))  int int4v;

#define WAVES 4
#define TS    16          // samples per wave tile
#define NJ    18          // skeleton joints
#define FLAT  576         // 18*32

#if defined(__HIP_DEVICE_COMPILE__) && __has_builtin(__builtin_amdgcn_tensor_load_to_lds)
#define USE_TDM 1
#else
#define USE_TDM 0
#endif

// LDS layout in half_t units
#define IMG_HALVES   (11 * 512)                 // A^T (2 imgs) + layer W (9 imgs)
#define BUFA_HALVES  (TS * 32 * 32)             // [s][f(<=32)][n32]
#define BUFB_HALVES  (TS * NJ * 32)             // [s][n18][f32]
#define XSTG_HALVES  (TS * NJ * 2 * 2)          // raw f32 X tile (2304 B / wave)
#define BUFA_BASE    IMG_HALVES
#define BUFB_BASE    (IMG_HALVES + WAVES * BUFA_HALVES)
#define XSTG_BASE    (BUFB_BASE + WAVES * BUFB_HALVES)
#define SMEM_HALVES  (XSTG_BASE + WAVES * XSTG_HALVES)
#define SMEM_BYTES   (SMEM_HALVES * 2)          // 225280 B < 320 KB WGP LDS

struct Params {
  const float* X; const float* A;
  const float* W[7]; const float* b[7];
  const float* Wout; const float* bout;
  float* out;
  int numTiles;
};

// A operand, 16 rows x K=32 f16, from an LDS matrix with 'rowStride' halves/row.
// Contiguous-k layout (same as B images): lanes 0-15 hold row M=lane, halves K0..15;
//             lanes 16-31 hold row M=lane-16, halves K16..31.
__device__ __forceinline__ v16h lds_load_a(const half_t* base, int rowStride,
                                           int rowBase, int lane) {
  int r    = rowBase + (lane & 15);
  int koff = (lane < 16) ? 0 : 16;
  const half_t* p = base + r * rowStride + koff;
  v8h lo = *(const v8h*)p;          // ds_load_b128
  v8h hi = *(const v8h*)(p + 8);    // ds_load_b128
  v16h o;
#pragma unroll
  for (int i = 0; i < 8; ++i) { o[i] = lo[i]; o[i + 8] = hi[i]; }
  return o;
}

// B operand from a pre-packed lane-order image (512 halves, 32B/lane).
__device__ __forceinline__ v16h lds_load_b(const half_t* img, int lane) {
  const half_t* p = img + lane * 16;
  v8h lo = *(const v8h*)p;
  v8h hi = *(const v8h*)(p + 8);
  v16h o;
#pragma unroll
  for (int i = 0; i < 8; ++i) { o[i] = lo[i]; o[i + 8] = hi[i]; }
  return o;
}

__global__ void __launch_bounds__(WAVES * 32, 1)
gcn_fused_wmma(Params P) {
  extern __shared__ __align__(16) unsigned char smem_raw[];
  half_t* smem = (half_t*)smem_raw;

  const int tid  = threadIdx.x;
  const int lane = tid & 31;
  const int wave = tid >> 5;

  // ---- pack shared B-operand images: img0/1 = A^T (N cols 0..15 / 16..31),
  //      img2.. = layer W^T tiles: L1..L5 one tile, L6/L7 two tiles each. ----
  {
    const int FinA[7]    = {2, 4, 8, 16, 16, 16, 32};
    const int FoutA[7]   = {4, 8, 16, 16, 16, 32, 32};
    const int layerOf[9] = {0, 1, 2, 3, 4, 5, 5, 6, 6};
    const int ntOf[9]    = {0, 0, 0, 0, 0, 0, 1, 0, 1};
    for (int idx = tid; idx < 11 * 512; idx += blockDim.x) {
      int img = idx >> 9, e = idx & 511;
      int l = e >> 4, j = e & 15;
      int k = (l < 16) ? j : 16 + j;       // K index of this half
      int n = (l < 16) ? l : l - 16;       // N index of this half
      float v = 0.0f;
      if (img < 2) {
        int col = img * 16 + n;
        if (k < NJ && col < NJ) v = P.A[col * NJ + k];   // B[k][n] = A[n][k]
      } else {
        int li = layerOf[img - 2], nt = ntOf[img - 2];
        int c = nt * 16 + n;
        if (k < FinA[li] && c < FoutA[li]) v = P.W[li][c * FinA[li] + k];
      }
      smem[idx] = (half_t)v;
    }
  }
  __syncthreads();

  half_t* bufA = smem + BUFA_BASE + wave * BUFA_HALVES;  // [s][f][n32]
  half_t* bufB = smem + BUFB_BASE + wave * BUFB_HALVES;  // [s][n18][f32]

  // bufA pad columns (addr&31 >= 18) are written by nobody but this memset and
  // the feature widths only grow layer to layer, so one zeroing lasts forever.
  {
    unsigned int* z = (unsigned int*)bufA;
    for (int idx = lane; idx < BUFA_HALVES / 2; idx += 32) z[idx] = 0u;
  }

  for (int tile = blockIdx.x * WAVES + wave; tile < P.numTiles;
       tile += gridDim.x * WAVES) {
    const int s0 = tile * TS;

#if USE_TDM
    // ---- Tensor Data Mover: DMA the raw f32 X tile (2304 B) into LDS ----
    {
      unsigned long long ga =
          (unsigned long long)P.X + (unsigned long long)s0 * (NJ * 2) * 4;
      uint4v g0;
      g0[0] = 1u;                                        // count=1, load, user
      g0[1] = (unsigned)(XSTG_BASE + wave * XSTG_HALVES) * 2u;  // lds byte addr
      g0[2] = (unsigned)(ga & 0xffffffffull);
      g0[3] = (unsigned)((ga >> 32) & 0x1ffffffull) | (2u << 30);  // type=image
      int8v g1 = (int8v)0;
      g1[0] = (int)(2u << 16);                 // data_size=4B, mask=0
      g1[1] = 0;                               // tensor_dim0 = 1<<30 (low bits 0)
      g1[2] = (int)(0x4000u | (1u << 16));     // tensor_dim0 hi, tensor_dim1=1
      g1[3] = (int)(576u << 16);               // tile_dim0 = 576 elements
      g1[4] = 0;                               // tile_dim1/2 unused
      g1[5] = 576;                             // tensor_dim0_stride
      int4v gz4 = (int4v)0;
#if __clang_major__ >= 23
      int8v gz8 = (int8v)0;
      __builtin_amdgcn_tensor_load_to_lds(g0, g1, gz4, gz4, gz8, 0);
#else
      __builtin_amdgcn_tensor_load_to_lds(g0, g1, gz4, gz4, 0);
#endif
      __builtin_amdgcn_s_wait_tensorcnt((short)0);
    }
#else
    // prefetch next tile's X through the gfx1250 prefetch path
    {
      int nxt = tile + (int)gridDim.x * WAVES;
      if (nxt < P.numTiles)
        __builtin_prefetch(P.X + (size_t)nxt * TS * NJ * 2 + lane * 16, 0, 1);
    }
#endif

    // zero bufB pad columns (layer 7 filled all 32 cols last iteration; widths
    // are non-decreasing within one iteration so once per tile suffices)
    {
      unsigned int* z = (unsigned int*)bufB;
      for (int idx = lane; idx < BUFB_HALVES / 2; idx += 32) z[idx] = 0u;
    }

    // ---- X tile -> bufA in layer-1 stage-A layout [s][f<2][n32] ----
#if USE_TDM
    {
      const float* xs = (const float*)(smem + XSTG_BASE + wave * XSTG_HALVES);
      for (int idx = lane; idx < TS * NJ * 2; idx += 32) {
        int s = idx / (NJ * 2), rem = idx - s * (NJ * 2);
        int n = rem >> 1, f = rem & 1;
        bufA[s * 64 + f * 32 + n] = (half_t)xs[idx];
      }
    }
#else
    for (int idx = lane; idx < TS * NJ * 2; idx += 32) {
      int s = idx / (NJ * 2), rem = idx - s * (NJ * 2);
      int n = rem >> 1, f = rem & 1;
      bufA[s * 64 + f * 32 + n] =
          (half_t)P.X[(size_t)(s0 + s) * (NJ * 2) + n * 2 + f];
    }
#endif

    // ------------------------------ 7 GCN layers ------------------------------
    for (int li = 0; li < 7; ++li) {
      const int lg2Fin  = (li == 0) ? 1 : ((li < 3) ? li + 1 : ((li < 6) ? 4 : 5));
      const int lg2Fout = (li < 2) ? li + 2 : ((li < 5) ? 4 : 5);
      const int Fin  = 1 << lg2Fin;
      const int Fout = 1 << lg2Fout;
      const int imgB = (li < 6) ? 2 + li : 9;
      const bool last = (li == 6);

      // ---- stage A: h[s,n,f] = sum_m X[s,m,f] * A[n,m]
      //      rows=(s,f) (16*Fin rows => Fin tiles), B = A^T images 0/1 ----
      for (int t = 0; t < Fin; ++t) {
        v16h a = lds_load_a(bufA, 32, t * 16, lane);
        v8f cz = {};
        v8f acc0 = __builtin_amdgcn_wmma_f32_16x16x32_f16(
            false, a, false, lds_load_b(smem, lane),       (short)0, cz, false, false);
        v8f acc1 = __builtin_amdgcn_wmma_f32_16x16x32_f16(
            false, a, false, lds_load_b(smem + 512, lane), (short)0, cz, false, false);
        v8h p0, p1;
#pragma unroll
        for (int i = 0; i < 8; ++i) { p0[i] = (half_t)acc0[i]; p1[i] = (half_t)acc1[i]; }
        int nl   = lane & 15;
        int mofs = (lane < 16) ? 0 : 8;
        int r0   = t * 16 + mofs;              // first M-row of this lane-half
        int s_   = r0 >> lg2Fin;               // constant across the 8 rows
        int f0   = r0 & (Fin - 1);             // multiple of 8 when Fin>=8
        bool hi  = (nl + 16 < NJ);
        if (Fin >= 8) {                        // 8 consecutive f, one sample
          half_t* d0 = bufB + s_ * FLAT + nl * 32 + f0;
          *(v8h*)d0 = p0;                      // ds_store_b128
          if (hi) *(v8h*)(bufB + s_ * FLAT + (nl + 16) * 32 + f0) = p1;
        } else {                               // Fin=2/4: contiguous runs
          for (int q = 0; q < 8; q += Fin) {
            half_t* d0 = bufB + s_ * FLAT + nl * 32;
            half_t* d1 = bufB + s_ * FLAT + (nl + 16) * 32;
            for (int e = 0; e < Fin; ++e) {
              d0[e] = p0[q + e];
              if (hi) d1[e] = p1[q + e];
            }
            ++s_;
          }
        }
      }

      // ---- stage B: out[s,n,o] = relu( sum_f h[s,n,f] * W[o,f] + b[o] )
      //      rows=(s,n) (288 rows => 18 tiles), B = W^T image(s) ----
      const int ntiles = (Fout + 15) >> 4;
      for (int t = 0; t < 18; ++t) {
        v16h a = lds_load_a(bufB, 32, t * 16, lane);
        for (int nt = 0; nt < ntiles; ++nt) {
          v8f cz = {};
          v8f acc = __builtin_amdgcn_wmma_f32_16x16x32_f16(
              false, a, false, lds_load_b(smem + (imgB + nt) * 512, lane),
              (short)0, cz, false, false);
          int c  = nt * 16 + (lane & 15);
          float bv = (c < Fout) ? P.b[li][c] : 0.0f;
          int mofs = (lane < 16) ? 0 : 8;
          int r0   = t * 16 + mofs;
          if (c < Fout) {                      // all-true when Fout>=16
            if (last) {
              // flat layout [s][n*32+c]: address is r*32 + c — pure stride 32
              half_t* d = bufB + r0 * 32 + c;
#pragma unroll
              for (int v = 0; v < 8; ++v) {
                *d = (half_t)fmaxf(acc[v] + bv, 0.0f);
                d += 32;
              }
            } else {
              // next stage-A layout [s][f=c][n]: n-stride 1 with wrap at 18
              int s_ = r0 / 18;
              int n_ = r0 - s_ * 18;
              half_t* d = bufA + s_ * (Fout * 32) + c * 32 + n_;
#pragma unroll
              for (int v = 0; v < 8; ++v) {
                *d = (half_t)fmaxf(acc[v] + bv, 0.0f);
                ++d; ++n_;
                if (n_ == 18) { n_ = 0; d += Fout * 32 - 18; }
              }
            }
          }
        }
      }
    }

    // -------- classifier: out[s,c] = flat[s,:] . Wout[c,:] + bout --------
    v8f acc[4] = {};
    for (int kc = 0; kc < 18; ++kc) {
      v16h a = lds_load_a(bufB + kc * 32, FLAT, 0, lane);  // rows = 16 samples
#pragma unroll
      for (int nt = 0; nt < 4; ++nt) {
        int c  = nt * 16 + (lane & 15);
        int ko = kc * 32 + ((lane < 16) ? 0 : 16);
        v16h bb;
        if (c < 60) {                     // Wout is 138 KB: L2-resident stream
          const float* p = P.Wout + (size_t)c * FLAT + ko;
#pragma unroll
          for (int i = 0; i < 16; ++i) bb[i] = (half_t)p[i];
        } else {
#pragma unroll
          for (int i = 0; i < 16; ++i) bb[i] = (half_t)0.0f;
        }
        acc[nt] = __builtin_amdgcn_wmma_f32_16x16x32_f16(
            false, a, false, bb, (short)0, acc[nt], false, false);
      }
    }
    {
      int mofs = (lane < 16) ? 0 : 8;
#pragma unroll
      for (int nt = 0; nt < 4; ++nt) {
        int c = nt * 16 + (lane & 15);
        if (c < 60) {
          float bv = P.bout[c];
#pragma unroll
          for (int v = 0; v < 8; ++v) {
            int s = v + mofs;
            P.out[(size_t)(s0 + s) * 60 + c] = acc[nt][v] + bv;  // 64B/half-wave
          }
        }
      }
    }
  }
}

extern "C" void kernel_launch(void* const* d_in, const int* in_sizes, int n_in,
                              void* d_out, int out_size, void* d_ws, size_t ws_size,
                              hipStream_t stream) {
  (void)n_in; (void)out_size; (void)d_ws; (void)ws_size;
  Params P;
  P.X = (const float*)d_in[0];
  P.A = (const float*)d_in[1];
  for (int i = 0; i < 7; ++i) {
    P.W[i] = (const float*)d_in[2 + 2 * i];
    P.b[i] = (const float*)d_in[3 + 2 * i];
  }
  P.Wout = (const float*)d_in[16];
  P.bout = (const float*)d_in[17];
  P.out  = (float*)d_out;

  int B = in_sizes[0] / (NJ * 2);       // 262144
  P.numTiles = B / TS;                  // 16384 wave tiles of 16 samples

  int blocks = (P.numTiles + WAVES - 1) / WAVES;
  if (blocks > 1024) blocks = 1024;
  if (blocks < 1) blocks = 1;

  gcn_fused_wmma<<<blocks, WAVES * 32, SMEM_BYTES, stream>>>(P);
}
